// QwenAttention_85830626443337
// MI455X (gfx1250) — matrix operand order repaired and checked by reading, not compiled, not run on hardware
//
#include <hip/hip_runtime.h>
#include <hip/hip_bf16.h>

typedef __bf16 bf16;
typedef __attribute__((ext_vector_type(16))) __bf16 v16bf;
typedef __attribute__((ext_vector_type(8)))  float  v8f;

union FragB { v16bf v; uint4 u[2]; };

#define T_SEQ 2048
#define QKVN  6144

// ---------------------------------------------------------------------------
// CDNA5 async global->LDS copy (ASYNCcnt-tracked), per-lane 16B.
// LDS byte address = low 32 bits of the generic pointer (ISA: LDS aperture
// addresses map to LDS by discarding the upper bits).
// ---------------------------------------------------------------------------
__device__ __forceinline__ void async_g2l_b128(void* lds_ptr, const void* gptr) {
  unsigned lds_off = (unsigned)(unsigned long long)lds_ptr;
  asm volatile("global_load_async_to_lds_b128 %0, %1, off"
               :: "v"(lds_off), "v"(gptr)
               : "memory");
}
__device__ __forceinline__ void wait_async0() {
  asm volatile("s_wait_asynccnt 0x0" ::: "memory");
}

// ---------------------------------------------------------------------------
// f32 -> bf16 matrix convert with destination leading-dim / column offset
// (lets us pack Wq|Wk|Wv into one 2048x6144 bf16 weight).
// ---------------------------------------------------------------------------
__global__ __launch_bounds__(256) void cvt_f32_bf16(
    const float* __restrict__ src, bf16* __restrict__ dst,
    int rows, int cols, int dst_ld, int col_off) {
  long i = (long)blockIdx.x * blockDim.x + threadIdx.x;
  long n = (long)rows * cols;
  if (i >= n) return;
  long r = i / cols, c = i % cols;
  dst[r * dst_ld + col_off + c] = (bf16)src[i];
}

// ---------------------------------------------------------------------------
// Generic GEMM: C[M,N] f32 = A[M,K] bf16 (row-major) @ B[K,N] bf16 (row-major)
// 128x128 block tile, k-step 32, 8 waves each computing 32x64 via 2x4 WMMA.
// A tile staged with global_load_async_to_lds_b128 (ASYNCcnt path);
// B tile staged via VGPR transpose into [n][k] so B-frags are contiguous.
// ---------------------------------------------------------------------------
__global__ __launch_bounds__(256) void gemm_bf16f32(
    const bf16* __restrict__ A, const bf16* __restrict__ B, float* __restrict__ C,
    int M, int N, int K) {
  __shared__ bf16 As[128][40];  // [m][k], stride 40 elems = 80B (16B aligned)
  __shared__ bf16 Bs[128][40];  // [n][k] (transposed on fill)

  const int tid = threadIdx.x;
  const int w = tid >> 5, l = tid & 31;
  const int wm = (w & 3) * 32;   // wave m-offset (4 waves along M)
  const int wn = (w >> 2) * 64;  // wave n-offset (2 waves along N)
  const int m0 = blockIdx.y * 128, n0 = blockIdx.x * 128;
  const int lh = l >> 4, lm = l & 15;

  v8f acc[2][4];
#pragma unroll
  for (int i = 0; i < 2; i++)
#pragma unroll
    for (int j = 0; j < 4; j++)
#pragma unroll
      for (int e = 0; e < 8; e++) acc[i][j][e] = 0.f;

  const int arow = tid >> 1, acolg = (tid & 1) * 16;  // A: 128 rows x 32 k
  const int brow = tid >> 3, bcolg = (tid & 7) * 16;  // B: 32 k rows x 128 n

  for (int k0 = 0; k0 < K; k0 += 32) {
    // stage A tile: async global->LDS, 2 x 16B per thread
    const bf16* ag = A + (long)(m0 + arow) * K + k0 + acolg;
    async_g2l_b128(&As[arow][acolg],     ag);
    async_g2l_b128(&As[arow][acolg + 8], ag + 8);
    // stage B tile transposed into Bs[n][k]
    const bf16* bg = B + (long)(k0 + brow) * N + n0 + bcolg;
    uint4 b0 = ((const uint4*)bg)[0], b1 = ((const uint4*)bg)[1];
    alignas(16) bf16 tmp[16];
    *(uint4*)&tmp[0] = b0; *(uint4*)&tmp[8] = b1;
#pragma unroll
    for (int j = 0; j < 16; j++) Bs[bcolg + j][brow] = tmp[j];
    wait_async0();     // wave's async copies landed in LDS
    __syncthreads();   // all waves' tiles visible

    if (k0 + 32 < K)
      __builtin_prefetch(A + (long)(m0 + arow) * K + k0 + 32 + acolg, 0, 0);

    // A-frag: lane holds row m, k = koff+{0..7} and koff+16+{0..7}, koff = (l>=16)?8:0
    const int koff = lh * 8;
    FragB a[2];
#pragma unroll
    for (int i = 0; i < 2; i++) {
      int m = wm + i * 16 + lm;
      a[i].u[0] = *(const uint4*)&As[m][koff];
      a[i].u[1] = *(const uint4*)&As[m][koff + 16];
    }
    // B-frag: lane holds col n, k = lh*8+{0..7} and lh*8+16+{0..7}
    FragB b[4];
#pragma unroll
    for (int j = 0; j < 4; j++) {
      int n = wn + j * 16 + lm;
      b[j].u[0] = *(const uint4*)&Bs[n][lh * 8];
      b[j].u[1] = *(const uint4*)&Bs[n][lh * 8 + 16];
    }
#pragma unroll
    for (int i = 0; i < 2; i++)
#pragma unroll
      for (int j = 0; j < 4; j++)
        acc[i][j] = __builtin_amdgcn_wmma_f32_16x16x32_bf16(
            false, a[i].v, false, b[j].v, (short)0, acc[i][j], false, false);
    __syncthreads();
  }

  // C layout: lane l, vgpr r -> (m = r + lh*8, n = lm)
#pragma unroll
  for (int i = 0; i < 2; i++)
#pragma unroll
    for (int j = 0; j < 4; j++)
#pragma unroll
      for (int r = 0; r < 8; r++) {
        int row = m0 + wm + i * 16 + r + lh * 8;
        int col = n0 + wn + j * 16 + lm;
        C[(long)row * N + col] = acc[i][j][r];
      }
}

// ---------------------------------------------------------------------------
// Fused RMSNorm + RoPE + scale + bf16 pack. One block per token.
// qkv row = [q:4096 | k:1024 | v:1024] f32.
// Outputs: Qb[32][T][128] (x 1/sqrt(128)), Kb[8][T][128], Vt[8][128][T].
// ---------------------------------------------------------------------------
__global__ __launch_bounds__(256) void norm_rope(
    const float* __restrict__ qkv, const float* __restrict__ qw,
    const float* __restrict__ kw, const float* __restrict__ sinp,
    const float* __restrict__ cosp, bf16* __restrict__ Qb,
    bf16* __restrict__ Kb, bf16* __restrict__ Vt) {
  __shared__ float row[QKVN];
  __shared__ float rstd[24];
  const int t = blockIdx.x;
  const int tid = threadIdx.x;
#pragma unroll
  for (int j = 0; j < 24; j++)
    row[tid + j * 256] = qkv[(long)t * QKVN + tid + j * 256];
  __syncthreads();

  const int w = tid >> 5, l = tid & 31;
  // 24 norm groups: 16 q-groups of 256, 8 k-groups of 128. Warp w -> groups w, w+8, w+16.
#pragma unroll
  for (int gi = 0; gi < 3; gi++) {
    int g = w + gi * 8;
    int sz = (g < 16) ? 256 : 128;
    int base = (g < 16) ? g * 256 : 4096 + (g - 16) * 128;
    float s = 0.f;
    for (int i = l; i < sz; i += 32) { float v = row[base + i]; s += v * v; }
#pragma unroll
    for (int msk = 16; msk >= 1; msk >>= 1) s += __shfl_xor(s, msk, 32);
    if (l == 0) rstd[g] = rsqrtf(s / (float)sz + 1e-6f);
  }
  __syncthreads();

  const float rscale = 0.08838834764831845f;  // 1/sqrt(128)
  // Q: 32 heads x 128
#pragma unroll
  for (int j = 0; j < 16; j++) {
    int o = tid + j * 256;
    int h = o >> 7, d = o & 127;
    int g = h >> 1;
    int inb = g * 256 + (h & 1) * 128;
    int wofs = (h & 1) * 128;
    int dd = (d < 64) ? d : d - 64;
    float rs = rstd[g];
    float x1 = row[inb + 2 * dd]     * rs * qw[wofs + 2 * dd];
    float x2 = row[inb + 2 * dd + 1] * rs * qw[wofs + 2 * dd + 1];
    float s = sinp[t * 64 + dd], c = cosp[t * 64 + dd];
    float out = (d < 64) ? (x1 * c - x2 * s) : (x1 * s + x2 * c);
    Qb[((long)h * T_SEQ + t) * 128 + d] = (bf16)(out * rscale);
  }
  // K: 8 heads x 128
#pragma unroll
  for (int j = 0; j < 4; j++) {
    int o = tid + j * 256;
    int h = o >> 7, d = o & 127;
    int inb = 4096 + h * 128;
    int dd = (d < 64) ? d : d - 64;
    float rs = rstd[16 + h];
    float x1 = row[inb + 2 * dd]     * rs * kw[2 * dd];
    float x2 = row[inb + 2 * dd + 1] * rs * kw[2 * dd + 1];
    float s = sinp[t * 64 + dd], c = cosp[t * 64 + dd];
    float out = (d < 64) ? (x1 * c - x2 * s) : (x1 * s + x2 * c);
    Kb[((long)h * T_SEQ + t) * 128 + d] = (bf16)out;
  }
  // V transposed: Vt[h][d][t]
#pragma unroll
  for (int j = 0; j < 4; j++) {
    int o = tid + j * 256;
    int h = o >> 7, d = o & 127;
    Vt[((long)h * 128 + d) * T_SEQ + t] = (bf16)row[5120 + o];
  }
}

// ---------------------------------------------------------------------------
// Flash-style attention. Grid (16 q-tiles, 32 q-heads), 256 threads (8 waves).
// Each wave owns 16 queries x 128 d. Full (non-causal) softmax over 2048 keys,
// key blocks of 128. Q/K/V fragments loaded straight from global (L2-resident);
// P transposed C-layout -> A-layout through a per-wave LDS pad.
// ---------------------------------------------------------------------------
__global__ __launch_bounds__(256) void attn_kernel(
    const bf16* __restrict__ Qb, const bf16* __restrict__ Kb,
    const bf16* __restrict__ Vt, bf16* __restrict__ Ob) {
  __shared__ bf16 Ps[8][16][128];  // per-wave P staging (row = m, col = k)
  const int qt = blockIdx.x;
  const int qh = blockIdx.y;
  const int kvh = qh >> 2;  // GQA: 4 query heads per kv head
  const int tid = threadIdx.x;
  const int w = tid >> 5, l = tid & 31;
  const int lh = l >> 4, lm = l & 15;
  const int qrow0 = qt * 128 + w * 16;

  const bf16* Qh = Qb + (long)qh * T_SEQ * 128;
  const bf16* Kh = Kb + (long)kvh * T_SEQ * 128;
  const bf16* Vh = Vt + (long)kvh * 128 * T_SEQ;

  // Resident Q A-fragments (4 k-steps over head dim 128)
  FragB qf[4];
#pragma unroll
  for (int ks = 0; ks < 4; ks++) {
    const bf16* p = Qh + (long)(qrow0 + lm) * 128 + ks * 32 + lh * 8;
    qf[ks].u[0] = *(const uint4*)p;
    qf[ks].u[1] = *(const uint4*)(p + 16);
  }

  float mrow[8], lrow[8];
  v8f O[8];
#pragma unroll
  for (int r = 0; r < 8; r++) {
    mrow[r] = -1e30f; lrow[r] = 0.f;
#pragma unroll
    for (int e = 0; e < 8; e++) O[r][e] = 0.f;
  }

  for (int kb = 0; kb < 16; kb++) {
    const int kk0 = kb * 128;
    // S = Q @ K^T : 8 key tiles of 16
    v8f S[8];
#pragma unroll
    for (int j = 0; j < 8; j++) {
#pragma unroll
      for (int e = 0; e < 8; e++) S[j][e] = 0.f;
#pragma unroll
      for (int ks = 0; ks < 4; ks++) {
        FragB b;
        const bf16* p = Kh + (long)(kk0 + j * 16 + lm) * 128 + ks * 32 + lh * 8;
        b.u[0] = *(const uint4*)p;
        b.u[1] = *(const uint4*)(p + 16);
        S[j] = __builtin_amdgcn_wmma_f32_16x16x32_bf16(
            false, qf[ks].v, false, b.v, (short)0, S[j], false, false);
      }
    }
    // Online softmax (rows live in 16-lane halves; reduce with xor 8..1)
    float mnew[8], alpha[8];
#pragma unroll
    for (int r = 0; r < 8; r++) {
      float mx = S[0][r];
#pragma unroll
      for (int j = 1; j < 8; j++) mx = fmaxf(mx, S[j][r]);
#pragma unroll
      for (int msk = 8; msk >= 1; msk >>= 1) mx = fmaxf(mx, __shfl_xor(mx, msk, 32));
      mnew[r] = fmaxf(mrow[r], mx);
      alpha[r] = __expf(mrow[r] - mnew[r]);
      mrow[r] = mnew[r];
    }
    float psum[8];
#pragma unroll
    for (int r = 0; r < 8; r++) psum[r] = 0.f;
#pragma unroll
    for (int j = 0; j < 8; j++)
#pragma unroll
      for (int r = 0; r < 8; r++) {
        float p = __expf(S[j][r] - mnew[r]);
        S[j][r] = p;
        psum[r] += p;
      }
#pragma unroll
    for (int r = 0; r < 8; r++) {
      float s = psum[r];
#pragma unroll
      for (int msk = 8; msk >= 1; msk >>= 1) s += __shfl_xor(s, msk, 32);
      lrow[r] = lrow[r] * alpha[r] + s;
#pragma unroll
      for (int j = 0; j < 8; j++) O[j][r] *= alpha[r];
    }
    // C-layout -> LDS (m, k): same-wave DS ops are in-order, no barrier needed
#pragma unroll
    for (int j = 0; j < 8; j++)
#pragma unroll
      for (int r = 0; r < 8; r++)
        Ps[w][r + lh * 8][j * 16 + lm] = (bf16)S[j][r];
    // O += P @ V : A-frags from LDS, B-frags from transposed V (contiguous keys)
#pragma unroll
    for (int ks = 0; ks < 4; ks++) {
      FragB pa;
      const bf16* pp = &Ps[w][lm][ks * 32 + lh * 8];
      pa.u[0] = *(const uint4*)pp;
      pa.u[1] = *(const uint4*)(pp + 16);
#pragma unroll
      for (int j = 0; j < 8; j++) {
        FragB b;
        const bf16* vp = Vh + (long)(j * 16 + lm) * T_SEQ + kk0 + ks * 32 + lh * 8;
        b.u[0] = *(const uint4*)vp;
        b.u[1] = *(const uint4*)(vp + 16);
        O[j] = __builtin_amdgcn_wmma_f32_16x16x32_bf16(
            false, pa.v, false, b.v, (short)0, O[j], false, false);
      }
    }
  }

  // Epilogue: O / rowsum, pack bf16 into attn_out[t][qh*128 + d]
#pragma unroll
  for (int r = 0; r < 8; r++) {
    float inv = 1.f / lrow[r];
    int qrow = qrow0 + r + lh * 8;
#pragma unroll
    for (int j = 0; j < 8; j++) {
      int col = qh * 128 + j * 16 + lm;
      Ob[(long)qrow * 4096 + col] = (bf16)(O[j][r] * inv);
    }
  }
}

// ---------------------------------------------------------------------------
// Host-side launch. Workspace layout (bytes):
//   [0,8M)    xb      2048x2048 bf16
//   [8M,32M)  wqkv    2048x6144 bf16  (Wq | Wk | Wv)
//   [32M,48M) wob     4096x2048 bf16
//   [48M,96M) qkvf    2048x6144 f32
//   [96M,112M)  Qb    32x2048x128 bf16
//   [112M,116M) Kb    8x2048x128 bf16
//   [116M,120M) Vt    8x128x2048 bf16
//   [120M,136M) attn  2048x4096 bf16
// ---------------------------------------------------------------------------
extern "C" void kernel_launch(void* const* d_in, const int* in_sizes, int n_in,
                              void* d_out, int out_size, void* d_ws, size_t ws_size,
                              hipStream_t stream) {
  const float* x  = (const float*)d_in[0];
  const float* Wq = (const float*)d_in[1];
  const float* Wk = (const float*)d_in[2];
  const float* Wv = (const float*)d_in[3];
  const float* Wo = (const float*)d_in[4];
  const float* qw = (const float*)d_in[5];
  const float* kw = (const float*)d_in[6];
  const float* sp = (const float*)d_in[7];
  const float* cp = (const float*)d_in[8];
  float* out = (float*)d_out;

  char* ws = (char*)d_ws;
  const size_t MB = (size_t)1 << 20;
  bf16*  xb   = (bf16*)(ws + 0);
  bf16*  wqkv = (bf16*)(ws + 8 * MB);
  bf16*  wob  = (bf16*)(ws + 32 * MB);
  float* qkvf = (float*)(ws + 48 * MB);
  bf16*  qb   = (bf16*)(ws + 96 * MB);
  bf16*  kb   = (bf16*)(ws + 112 * MB);
  bf16*  vt   = (bf16*)(ws + 116 * MB);
  bf16*  ab   = (bf16*)(ws + 120 * MB);

  auto cvt = [&](const float* s, bf16* dstp, int r, int c, int ld, int off) {
    long n = (long)r * c;
    cvt_f32_bf16<<<(unsigned)((n + 255) / 256), 256, 0, stream>>>(s, dstp, r, c, ld, off);
  };
  cvt(x,  xb,   2048, 2048, 2048, 0);
  cvt(Wq, wqkv, 2048, 4096, 6144, 0);
  cvt(Wk, wqkv, 2048, 1024, 6144, 4096);
  cvt(Wv, wqkv, 2048, 1024, 6144, 5120);
  cvt(Wo, wob,  4096, 2048, 2048, 0);

  gemm_bf16f32<<<dim3(48, 16), 256, 0, stream>>>(xb, wqkv, qkvf, 2048, 6144, 2048);
  norm_rope<<<2048, 256, 0, stream>>>(qkvf, qw, kw, sp, cp, qb, kb, vt);
  attn_kernel<<<dim3(16, 32), 256, 0, stream>>>(qb, kb, vt, ab);
  gemm_bf16f32<<<dim3(16, 16), 256, 0, stream>>>(ab, wob, out, 2048, 2048, 4096);
}
